// GCNPredictor_31284541784068
// MI455X (gfx1250) — matrix operand order repaired and checked by reading, not compiled, not run on hardware
//
#include <hip/hip_runtime.h>
#include <math.h>

// ---------------- problem constants ----------------
#define BX    8
#define NN    2000
#define CC    4096
#define HIDD  42
#define NCC   20
#define NP    2048          // padded N (multiple of 128)
#define HP    48            // padded hidden (3 x 16)
#define CP    32            // padded classes (2 x 16)
#define LDSTR 40            // LDS row stride in halves (32 + 8 pad)
#define TSTR  136           // LDS transpose-tile stride in halves (16B-aligned rows)
#define NTI   (NP / 128)    // 16 row-tiles per image

typedef __attribute__((ext_vector_type(16))) _Float16 v16h;
typedef __attribute__((ext_vector_type(8)))  float    v8f;
typedef int v4i __attribute__((vector_size(16)));

struct __align__(16) F4 { float x, y, z, w; };   // POD 16B vector (union-safe)

union V8F { v8f v; float f[8]; };
union H8  { _Float16 e[8]; F4 f4; };             // 8 halves == 16 bytes

// ---------------- CDNA5 helpers ----------------
__device__ __forceinline__ v8f wmma16(v16h a, v16h b, v8f c) {
  // D = A(16x32 f16) * B(32x16 f16) + C(16x16 f32)
  return __builtin_amdgcn_wmma_f32_16x16x32_f16(
      false, a, false, b, (short)0, c, false, false);
}

__device__ __forceinline__ void wait_async0() {
  asm volatile("s_wait_asynccnt 0x0" ::: "memory");
}

// async 16B global -> LDS copy (ASYNCcnt-tracked); fallback: VGPR copy
__device__ __forceinline__ void async_copy16(const _Float16* g, _Float16* l) {
#if defined(__gfx1250__) && __has_builtin(__builtin_amdgcn_global_load_async_to_lds_b128)
  using gp4_t = __attribute__((address_space(1))) v4i*;
  using lp4_t = __attribute__((address_space(3))) v4i*;
  __builtin_amdgcn_global_load_async_to_lds_b128((gp4_t)g, (lp4_t)l, 0, 0);
#else
  *(F4*)l = *(const F4*)g;
#endif
}

// A operand, 16x32 f16.  Lane L: M = L&15.  Halves 0..7 -> K = kb..kb+7,
// halves 8..15 -> K = 16+kb..16+kb+7, kb = (L>>4)*8  (ISA 7.12.2 table).
__device__ __forceinline__ v16h frag_a(const _Float16* lds, int rowBase, int lane) {
  const int m = lane & 15, kb = (lane >> 4) << 3;
  const _Float16* p = lds + (rowBase + m) * LDSTR + kb;
  v16h r;
  ((F4*)&r)[0] = *(const F4*)(p);
  ((F4*)&r)[1] = *(const F4*)(p + 16);
  return r;
}

// B operand, 32x16 f16 (KxN).  Lane L: N = L&15, halves 0..7 -> K = kb..kb+7,
// halves 8..15 -> K = 16+kb..16+kb+7, kb = (L>>4)*8.  LDS holds tile as [n][k].
__device__ __forceinline__ v16h frag_b(const _Float16* lds, int colBase, int lane) {
  const int n = lane & 15, kb = (lane >> 4) << 3;
  const _Float16* p = lds + (colBase + n) * LDSTR + kb;
  v16h r;
  ((F4*)&r)[0] = *(const F4*)(p);
  ((F4*)&r)[1] = *(const F4*)(p + 16);
  return r;
}

// load 16 fp32 (4 x b128) into registers
__device__ __forceinline__ void gload16(const float* base, int kb, int sseg, F4* r) {
  const F4* p = (const F4*)(base + kb) + sseg * 4;
  r[0] = p[0]; r[1] = p[1]; r[2] = p[2]; r[3] = p[3];
}

// convert 16 scaled fp32 regs -> 16 f16 halves, store as 2 x b128 into LDS
__device__ __forceinline__ void cvt_store16(const F4* r, float sc, _Float16* dst) {
  H8 lo, hi;
  lo.e[0] = (_Float16)(r[0].x * sc); lo.e[1] = (_Float16)(r[0].y * sc);
  lo.e[2] = (_Float16)(r[0].z * sc); lo.e[3] = (_Float16)(r[0].w * sc);
  lo.e[4] = (_Float16)(r[1].x * sc); lo.e[5] = (_Float16)(r[1].y * sc);
  lo.e[6] = (_Float16)(r[1].z * sc); lo.e[7] = (_Float16)(r[1].w * sc);
  hi.e[0] = (_Float16)(r[2].x * sc); hi.e[1] = (_Float16)(r[2].y * sc);
  hi.e[2] = (_Float16)(r[2].z * sc); hi.e[3] = (_Float16)(r[2].w * sc);
  hi.e[4] = (_Float16)(r[3].x * sc); hi.e[5] = (_Float16)(r[3].y * sc);
  hi.e[6] = (_Float16)(r[3].z * sc); hi.e[7] = (_Float16)(r[3].w * sc);
  *(F4*)dst       = lo.f4;
  *(F4*)(dst + 8) = hi.f4;
}

// ---------------- 1. per-row stats: inv_sum (row-normalize), inv_norm (cosine) --
__global__ __launch_bounds__(256) void k_stats(const float* __restrict__ x,
                                               float* __restrict__ scales) {
  __shared__ float rs[256], rss[256];
  const float* xr = x + (size_t)blockIdx.x * CC;
  const int t = threadIdx.x;
  float s = 0.f, ss = 0.f;
  for (int c = t; c < CC; c += 256) { float v = xr[c]; s += v; ss += v * v; }
  rs[t] = s; rss[t] = ss; __syncthreads();
  for (int off = 128; off; off >>= 1) {
    if (t < off) { rs[t] += rs[t + off]; rss[t] += rss[t + off]; }
    __syncthreads();
  }
  if (t == 0) {
    float sum = rs[0];
    scales[(size_t)blockIdx.x * 2]     = (sum == 0.f) ? 0.f : 1.f / sum;
    scales[(size_t)blockIdx.x * 2 + 1] = 1.f / fmaxf(sqrtf(rss[0]), 1e-12f);
  }
}

// ---------------- 2. transpose W1 (so K is lane-contiguous in k_xw) -------------
__global__ void k_wt(const float* __restrict__ Wc, const float* __restrict__ Wd,
                     float* __restrict__ Wtc, float* __restrict__ Wtd) {
  int idx = blockIdx.x * blockDim.x + threadIdx.x;
  if (idx >= CC * HIDD) return;
  int c = idx / HIDD, h = idx - c * HIDD;
  const float* W = blockIdx.y ? Wd : Wc;
  float* Wt      = blockIdx.y ? Wtd : Wtc;
  Wt[(size_t)h * CC + c] = W[idx];
}

// ---------------- 3. IoU binary adjacency into padded [NP x NP] f16 -------------
__global__ __launch_bounds__(256) void k_iou(const float* __restrict__ boxes,
                                             _Float16* __restrict__ Aadj) {
  size_t idx = (size_t)blockIdx.x * 256 + threadIdx.x;   // b*NP*NP + i*NP + j
  int j = (int)(idx & (NP - 1));
  size_t r = idx >> 11;
  int i = (int)(r & (NP - 1));
  int b = (int)(r >> 11);
  _Float16 outv = (_Float16)0.f;
  if (i < NN && j < NN) {
    const float* bi = boxes + ((size_t)b * NN + i) * 4;
    const float* bj = boxes + ((size_t)b * NN + j) * 4;
    float aa = (bi[2] - bi[0]) * (bi[3] - bi[1]);
    float ab = (bj[2] - bj[0]) * (bj[3] - bj[1]);
    float iw = fminf(bi[2], bj[2]) - fmaxf(bi[0], bj[0]); iw = iw > 0.f ? iw : 0.f;
    float ih = fminf(bi[3], bj[3]) - fmaxf(bi[1], bj[1]); ih = ih > 0.f ? ih : 0.f;
    float inter = iw * ih;
    float iou = inter / fmaxf(aa + ab - inter, 1e-12f);
    if (iou >= 0.5f) outv = (_Float16)1.f;
  }
  Aadj[idx] = outv;
}

// ---------------- 4. Gram + threshold (WMMA, pipelined, symmetric tiles) --------
__global__ __launch_bounds__(256) void k_gram(const float* __restrict__ x,
                                              const float* __restrict__ scales,
                                              _Float16* __restrict__ Asim) {
  __shared__ _Float16 smem[20480];   // 4 x (128*LDSTR) ping-pong; reused as 128xTSTR tile
  // layout: [0,5120) = A buf0, [5120,10240) = A buf1,
  //         [10240,15360) = B buf0, [15360,20480) = B buf1

  const int b = blockIdx.y;
  int tl = blockIdx.x, ti = 0;
  while (tl >= (NTI - ti)) { tl -= (NTI - ti); ++ti; }   // upper-triangle tile map
  const int tj = ti + tl;

  const int t = threadIdx.x, wid = t >> 5, lane = t & 31;
  const int mh = wid >> 2, nq = wid & 3;                 // wave grid 2(M) x 4(N)
  const int srow = t >> 1, sseg = t & 1;                 // staging: 128 rows x 2 segs

  const int rA = ti * 128 + srow, rB = tj * 128 + srow;
  const int gA = rA < NN ? rA : NN - 1, gB = rB < NN ? rB : NN - 1;
  const float scA = (rA < NN) ? scales[((size_t)b * NN + gA) * 2 + 1] : 0.f;
  const float scB = (rB < NN) ? scales[((size_t)b * NN + gB) * 2 + 1] : 0.f;
  const float* baseA = x + ((size_t)b * NN + gA) * CC;
  const float* baseB = x + ((size_t)b * NN + gB) * CC;
  const int soff = srow * LDSTR + sseg * 16;

  const v8f z8 = {0.f, 0.f, 0.f, 0.f, 0.f, 0.f, 0.f, 0.f};
  V8F acc[4][2];
#pragma unroll
  for (int i = 0; i < 4; ++i)
#pragma unroll
    for (int j = 0; j < 2; ++j) acc[i][j].v = z8;

  // prime buffer 0
  F4 ra[4], rb[4];
  gload16(baseA, 0, sseg, ra);
  gload16(baseB, 0, sseg, rb);
  cvt_store16(ra, scA, smem + soff);
  cvt_store16(rb, scB, smem + 10240 + soff);
  __syncthreads();

  int p = 0;
  for (int kb = 0; kb < CC; kb += 32) {
    const bool more = (kb + 32) < CC;
    if (more) {                             // issue next-chunk loads early
      gload16(baseA, kb + 32, sseg, ra);
      gload16(baseB, kb + 32, sseg, rb);
    }
    const _Float16* curA = smem + p * 5120;          // runtime ptr math (no
    const _Float16* curB = smem + 10240 + p * 5120;  //  static addrspacecast)
    v16h bf0 = frag_b(curB, nq * 32, lane);
    v16h bf1 = frag_b(curB, nq * 32 + 16, lane);
#pragma unroll
    for (int mt = 0; mt < 4; ++mt) {
      v16h af = frag_a(curA, mh * 64 + mt * 16, lane);
      acc[mt][0].v = wmma16(af, bf0, acc[mt][0].v);
      acc[mt][1].v = wmma16(af, bf1, acc[mt][1].v);
    }
    if (more) {                             // convert + stage into other buffer
      cvt_store16(ra, scA, smem + (p ^ 1) * 5120 + soff);
      cvt_store16(rb, scB, smem + 10240 + (p ^ 1) * 5120 + soff);
    }
    __syncthreads();                        // single barrier per K-step
    p ^= 1;
  }

  // epilogue: threshold -> LDS tile -> fully coalesced b128 stores (both mirrors)
  _Float16* T = smem;                       // 128 x TSTR (17408 halves <= 20480)
  _Float16* Ab = Asim + (size_t)b * NP * NP;
  const int mrow = t >> 1, mseg = t & 1;

  // pass 1: natural orientation -> tile (ti, tj)
#pragma unroll
  for (int mt = 0; mt < 4; ++mt)
#pragma unroll
    for (int nt = 0; nt < 2; ++nt) {
      const int ln = nq * 32 + nt * 16 + (lane & 15);
#pragma unroll
      for (int r = 0; r < 8; ++r) {
        const int lm = mh * 64 + mt * 16 + r + ((lane >> 4) << 3);
        T[lm * TSTR + ln] = (acc[mt][nt].f[r] >= 0.5f) ? (_Float16)1.f : (_Float16)0.f;
      }
    }
  __syncthreads();
  {
    const F4* src = (const F4*)&T[mrow * TSTR + mseg * 64];
    F4* dst = (F4*)&Ab[(size_t)(ti * 128 + mrow) * NP + tj * 128 + mseg * 64];
#pragma unroll
    for (int q = 0; q < 8; ++q) dst[q] = src[q];
  }
  __syncthreads();
  // pass 2: transposed -> tile (tj, ti)   (sim is symmetric)
#pragma unroll
  for (int mt = 0; mt < 4; ++mt)
#pragma unroll
    for (int nt = 0; nt < 2; ++nt) {
      const int ln = nq * 32 + nt * 16 + (lane & 15);
#pragma unroll
      for (int r = 0; r < 8; ++r) {
        const int lm = mh * 64 + mt * 16 + r + ((lane >> 4) << 3);
        T[ln * TSTR + lm] = (acc[mt][nt].f[r] >= 0.5f) ? (_Float16)1.f : (_Float16)0.f;
      }
    }
  __syncthreads();
  {
    const F4* src = (const F4*)&T[mrow * TSTR + mseg * 64];
    F4* dst = (F4*)&Ab[(size_t)(tj * 128 + mrow) * NP + ti * 128 + mseg * 64];
#pragma unroll
    for (int q = 0; q < 8; ++q) dst[q] = src[q];
  }
}

// ---------------- 5. XW = row_normalize(x) @ W1  (fp32 wave reductions) ---------
__global__ __launch_bounds__(256) void k_xw(const float* __restrict__ x,
                                            const float* __restrict__ scales,
                                            const float* __restrict__ Wtc,
                                            const float* __restrict__ Wtd,
                                            _Float16* __restrict__ XWc,
                                            _Float16* __restrict__ XWd) {
  __shared__ float lx[CC];
  const int blk = blockIdx.x;                 // b*NP + row
  const int b = blk >> 11, row = blk & (NP - 1);
  const int t = threadIdx.x, wid = t >> 5, lane = t & 31;
  if (row >= NN) {                            // zero pad rows (uniform branch)
    if (t < HP) {
      XWc[((size_t)b * NP + row) * HP + t] = (_Float16)0.f;
      XWd[((size_t)b * NP + row) * HP + t] = (_Float16)0.f;
    }
    return;
  }
  const float* xr = x + ((size_t)b * NN + row) * CC;
  for (int v = t; v < CC / 4; v += 256) ((F4*)lx)[v] = ((const F4*)xr)[v];
  __syncthreads();
  const float inv_s = scales[((size_t)b * NN + row) * 2];
  for (int it = 0; it < 12; ++it) {
    const int o = wid + (it << 3);            // 0..95: [cls 0..47 | det 48..95]
    const int g = o / HP, col = o - g * HP;
    float r = 0.f;
    if (col < HIDD) {
      const float* wcol = (g ? Wtd : Wtc) + (size_t)col * CC;
      for (int c = lane; c < CC; c += 32) r += lx[c] * wcol[c];
      for (int m = 16; m; m >>= 1) r += __shfl_xor(r, m, 32);
      r *= inv_s;
    }
    if (lane == 0) {
      size_t oidx = ((size_t)b * NP + row) * HP + col;
      if (g == 0) XWc[oidx] = (_Float16)r; else XWd[oidx] = (_Float16)r;
    }
  }
}

// ---------------- 6. Aggregation  out = A_bin @ V (WMMA, async LDS staging) -----
__global__ __launch_bounds__(256) void k_agg(const _Float16* __restrict__ A,
                                             const _Float16* __restrict__ V,
                                             const float* __restrict__ bias,
                                             int bias_len, int ncols, int do_relu,
                                             _Float16* __restrict__ outH,
                                             float* __restrict__ outF) {
  __shared__ _Float16 lAg[2][128 * LDSTR];
  __shared__ _Float16 lVg[2][HP * LDSTR];
  const int b = blockIdx.y, Rblk = blockIdx.x * 128;
  const int t = threadIdx.x, wid = t >> 5, lane = t & 31;
  const int srow = t >> 1, sseg = t & 1;
  const int nt2 = ncols >> 4;                       // 3 (HP) or 2 (CP)
  const int ecnt = (32 * ncols) >> 8;               // V halves per thread (6 or 4)
  const v8f z8 = {0.f, 0.f, 0.f, 0.f, 0.f, 0.f, 0.f, 0.f};
  V8F acc[3];
  acc[0].v = z8; acc[1].v = z8; acc[2].v = z8;

  const _Float16* Arow = A + ((size_t)b * NP + Rblk + srow) * NP;
  const int aoff = srow * LDSTR + sseg * 16;

  unsigned short vr[6];

  // prime buffer 0
  async_copy16(Arow + sseg * 16,     &lAg[0][aoff]);
  async_copy16(Arow + sseg * 16 + 8, &lAg[0][aoff + 8]);
  for (int i = 0; i < ecnt; ++i) {
    int e = t + (i << 8), k = e / ncols, n = e - k * ncols;
    vr[i] = *(const unsigned short*)&V[((size_t)b * NP + k) * ncols + n];
  }
  for (int i = 0; i < ecnt; ++i) {
    int e = t + (i << 8), k = e / ncols, n = e - k * ncols;
    *(unsigned short*)&lVg[0][n * LDSTR + k] = vr[i];
  }
  wait_async0();
  __syncthreads();

  int p = 0;
  for (int kb = 0; kb < NP; kb += 32) {
    const bool more = (kb + 32) < NP;
    if (more) {   // async A -> other buffer + V loads in flight during WMMAs
      async_copy16(Arow + kb + 32 + sseg * 16,     &lAg[p ^ 1][aoff]);
      async_copy16(Arow + kb + 32 + sseg * 16 + 8, &lAg[p ^ 1][aoff + 8]);
      for (int i = 0; i < ecnt; ++i) {
        int e = t + (i << 8), k = e / ncols, n = e - k * ncols;
        vr[i] = *(const unsigned short*)&V[((size_t)b * NP + kb + 32 + k) * ncols + n];
      }
    }
    v16h af = frag_a(lAg[p], wid * 16, lane);
    for (int nt = 0; nt < nt2; ++nt) {
      v16h bf = frag_b(lVg[p], nt * 16, lane);
      acc[nt].v = wmma16(af, bf, acc[nt].v);
    }
    if (more) {
      for (int i = 0; i < ecnt; ++i) {
        int e = t + (i << 8), k = e / ncols, n = e - k * ncols;
        *(unsigned short*)&lVg[p ^ 1][n * LDSTR + k] = vr[i];
      }
    }
    wait_async0();
    __syncthreads();
    p ^= 1;
  }

  for (int nt = 0; nt < nt2; ++nt) {
    const int col = nt * 16 + (lane & 15);
    const float bb = (col < bias_len) ? bias[col] : 0.f;
#pragma unroll
    for (int r = 0; r < 8; ++r) {
      const int gRow = Rblk + wid * 16 + r + ((lane >> 4) << 3);
      const float vv = acc[nt].f[r] + bb;
      const size_t o = ((size_t)b * NP + gRow) * ncols + col;
      if (do_relu) outH[o] = (_Float16)(vv > 0.f ? vv : 0.f);
      else         outF[o] = vv;
    }
  }
}

// ---------------- 7. HW2 = h @ W2 (42x20, fp32) ---------------------------------
__global__ __launch_bounds__(256) void k_hw2(const _Float16* __restrict__ Hc,
                                             const _Float16* __restrict__ Hd,
                                             const float* __restrict__ W2c,
                                             const float* __restrict__ W2d,
                                             _Float16* __restrict__ Oc,
                                             _Float16* __restrict__ Od) {
  const int idx = blockIdx.x * 256 + threadIdx.x;  // (b*NP+row)*CP + col
  const int col = idx & (CP - 1);
  const int row = (idx >> 5) & (NP - 1);
  const int b = idx >> 16;
  const int g = blockIdx.y;
  const _Float16* H = g ? Hd : Hc;
  const float* W2   = g ? W2d : W2c;
  _Float16* O       = g ? Od : Oc;
  float s = 0.f;
  if (col < NCC) {
    const _Float16* hr = H + ((size_t)b * NP + row) * HP;
    for (int h = 0; h < HIDD; ++h) s += (float)hr[h] * W2[h * NCC + col];
  }
  O[(size_t)idx] = (_Float16)s;
}

// ---------------- 8a. softmax over classes (one wave per proposal) --------------
__global__ __launch_bounds__(256) void k_softmax_cls(const float* __restrict__ logits,
                                                     float* __restrict__ out) {
  const int t = threadIdx.x, wid = t >> 5, lane = t & 31;
  const int rowIdx = blockIdx.x * 8 + wid;          // < B*NN
  const int b = rowIdx / NN, r = rowIdx - b * NN;
  float v = (lane < NCC) ? logits[((size_t)b * NP + r) * CP + lane] : -3.4e38f;
  float m = v;
  for (int k = 16; k; k >>= 1) m = fmaxf(m, __shfl_xor(m, k, 32));
  float e = (lane < NCC) ? __expf(v - m) : 0.f;
  float s = e;
  for (int k = 16; k; k >>= 1) s += __shfl_xor(s, k, 32);
  if (lane < NCC) out[(size_t)rowIdx * NCC + lane] = e / s;
}

// ---------------- 8b. softmax over proposals (one block per (b, class)) ---------
__global__ __launch_bounds__(256) void k_softmax_det(const float* __restrict__ logits,
                                                     float* __restrict__ out) {
  __shared__ float red[256];
  const int t = threadIdx.x;
  const int b = blockIdx.x / NCC, c = blockIdx.x - b * NCC;
  const float* base = logits + (size_t)b * NP * CP + c;
  float m = -3.4e38f;
  for (int r = t; r < NN; r += 256) m = fmaxf(m, base[(size_t)r * CP]);
  red[t] = m; __syncthreads();
  for (int off = 128; off; off >>= 1) {
    if (t < off) red[t] = fmaxf(red[t], red[t + off]);
    __syncthreads();
  }
  m = red[0]; __syncthreads();
  float s = 0.f;
  for (int r = t; r < NN; r += 256) s += __expf(base[(size_t)r * CP] - m);
  red[t] = s; __syncthreads();
  for (int off = 128; off; off >>= 1) {
    if (t < off) red[t] += red[t + off];
    __syncthreads();
  }
  const float inv = 1.f / red[0];
  for (int r = t; r < NN; r += 256)
    out[(size_t)(b * NN + r) * NCC + c] = __expf(base[(size_t)r * CP] - m) * inv;
}

// ---------------- launch --------------------------------------------------------
extern "C" void kernel_launch(void* const* d_in, const int* in_sizes, int n_in,
                              void* d_out, int out_size, void* d_ws, size_t ws_size,
                              hipStream_t stream) {
  const float* x1     = (const float*)d_in[0];
  // d_in[1] (x2) is unused by the reference after pooling
  const float* boxes  = (const float*)d_in[2];
  const float* W_cls1 = (const float*)d_in[3];
  const float* b_cls1 = (const float*)d_in[4];
  const float* W_cls2 = (const float*)d_in[5];
  const float* b_cls2 = (const float*)d_in[6];
  const float* W_det1 = (const float*)d_in[7];
  const float* b_det1 = (const float*)d_in[8];
  const float* W_det2 = (const float*)d_in[9];
  const float* b_det2 = (const float*)d_in[10];
  float* out = (float*)d_out;

  char* w = (char*)d_ws;                                      // all sizes 256B-aligned
  float* scales   = (float*)w;     w += (size_t)BX * NN * 2 * sizeof(float);
  _Float16* Aadj  = (_Float16*)w;  w += (size_t)BX * NP * NP * sizeof(_Float16);
  _Float16* Asim  = (_Float16*)w;  w += (size_t)BX * NP * NP * sizeof(_Float16);
  _Float16* XWc   = (_Float16*)w;  w += (size_t)BX * NP * HP * sizeof(_Float16);
  _Float16* XWd   = (_Float16*)w;  w += (size_t)BX * NP * HP * sizeof(_Float16);
  _Float16* Hc    = (_Float16*)w;  w += (size_t)BX * NP * HP * sizeof(_Float16);
  _Float16* Hd    = (_Float16*)w;  w += (size_t)BX * NP * HP * sizeof(_Float16);
  _Float16* HW2c  = (_Float16*)w;  w += (size_t)BX * NP * CP * sizeof(_Float16);
  _Float16* HW2d  = (_Float16*)w;  w += (size_t)BX * NP * CP * sizeof(_Float16);
  float* logC     = (float*)w;     w += (size_t)BX * NP * CP * sizeof(float);
  float* logD     = (float*)w;     w += (size_t)BX * NP * CP * sizeof(float);
  float* Wtc      = (float*)w;     w += (size_t)HIDD * CC * sizeof(float);
  float* Wtd      = (float*)w;     w += (size_t)HIDD * CC * sizeof(float);

  k_stats<<<BX * NN, 256, 0, stream>>>(x1, scales);
  k_wt<<<dim3((CC * HIDD + 255) / 256, 2), 256, 0, stream>>>(W_cls1, W_det1, Wtc, Wtd);
  k_iou<<<(BX * NP * NP) / 256, 256, 0, stream>>>(boxes, Aadj);
  k_gram<<<dim3(NTI * (NTI + 1) / 2, BX), 256, 0, stream>>>(x1, scales, Asim);
  k_xw<<<BX * NP, 256, 0, stream>>>(x1, scales, Wtc, Wtd, XWc, XWd);

  k_agg<<<dim3(NP / 128, BX), 256, 0, stream>>>(Aadj, XWc, b_cls1, HIDD, HP, 1, Hc, nullptr);
  k_agg<<<dim3(NP / 128, BX), 256, 0, stream>>>(Asim, XWd, b_det1, HIDD, HP, 1, Hd, nullptr);
  k_hw2<<<dim3((BX * NP * CP) / 256, 2), 256, 0, stream>>>(Hc, Hd, W_cls2, W_det2, HW2c, HW2d);
  k_agg<<<dim3(NP / 128, BX), 256, 0, stream>>>(Aadj, HW2c, b_cls2, NCC, CP, 0, nullptr, logC);
  k_agg<<<dim3(NP / 128, BX), 256, 0, stream>>>(Asim, HW2d, b_det2, NCC, CP, 0, nullptr, logD);

  k_softmax_cls<<<(BX * NN) / 8, 256, 0, stream>>>(logC, out);
  k_softmax_det<<<BX * NCC, 256, 0, stream>>>(logD, out + (size_t)BX * NN * NCC);

  (void)in_sizes; (void)n_in; (void)out_size; (void)ws_size;
}
